// Net_10273561772481
// MI455X (gfx1250) — matrix operand order repaired and checked by reading, compiled, not run on hardware
//
#include <hip/hip_runtime.h>
#include <hip/hip_bf16.h>

#define NN    100000
#define EE    1600000
#define F_IN  512
#define HH    64
#define RR    2
#define BB    30
#define NCOLS 192   // [ h_r0 (64) | h_r1 (64) | x@root (64) ]
#define M_BLK 32    // 100000 = 3125 * 32, no tail -> EXEC stays all-ones for WMMA

typedef __bf16 bf16_t;
typedef __attribute__((ext_vector_type(16))) __bf16 v16bf;
typedef __attribute__((ext_vector_type(8)))  __bf16 v8bf;
typedef __attribute__((ext_vector_type(4)))  __bf16 v4bf;
typedef __attribute__((ext_vector_type(8)))  float  v8f;

__device__ __forceinline__ float leaky02(float x) { return x > 0.f ? x : 0.2f * x; }

__device__ __forceinline__ void atomicMaxFloat(float* addr, float val) {
  if (val >= 0.f) atomicMax((int*)addr, __float_as_int(val));
  else            atomicMin((unsigned int*)addr, __float_as_uint(val));
}

// ---------------------------------------------------------------------------
// Fold basis decomposition + root into a k-major bf16 weight table:
//   Wt[col][k], col 0..127 -> W_r[k, col&63] = sum_b comp[r,b]*basis[b,k,col&63]
//               col 128..191 -> root[k, col-128]
// k-major layout => each WMMA lane reads 32 contiguous bytes for its B frag.
// ---------------------------------------------------------------------------
template <int K>
__global__ void k_prep_w(const float* __restrict__ basis, const float* __restrict__ comp,
                         const float* __restrict__ root, bf16_t* __restrict__ Wt) {
  int idx = blockIdx.x * blockDim.x + threadIdx.x;
  if (idx >= NCOLS * K) return;
  int col = idx / K, k = idx % K;
  float v;
  if (col < 2 * HH) {
    int r = col >> 6, o = col & 63;
    float s = 0.f;
#pragma unroll
    for (int b = 0; b < BB; ++b)
      s += comp[r * BB + b] * basis[((size_t)b * K + k) * HH + o];
    v = s;
  } else {
    v = root[(size_t)k * HH + (col - 2 * HH)];
  }
  Wt[(size_t)col * K + k] = (bf16_t)v;
}

// ---------------------------------------------------------------------------
// C[N x 192] = A[N x K] (fp32 -> bf16 staged in LDS) @ Wt^T (bf16, k-major)
// One block = 32 output rows; 6 waves. Each wave owns a 32x16 C strip for
// two n-tiles (nt = wave, wave+6): 4 accumulators. Per k-step: 4 ds_load_b128
// (two A frags, shared across the 2 n-tiles) + 2 global B frags (each reused
// by 2 WMMAs) + 4x v_wmma_f32_16x16x32_bf16.
// Fragment layouts used here (A/B: contiguous k order, same for both, not the ISA table order):
//   A (16x32 bf16): M = lane&15; k-half by lane[4]; elem e -> k = kb+16*hi+e
//   B (32x16 bf16): N = lane&15; elem e -> k = kb + 16*hi + e (contiguous in
//                   the k-major Wt row -> one 32B load)
//   C/D f32: vgpr v -> M = v + 8*hi, N = lane&15
// ---------------------------------------------------------------------------
template <int K>
__global__ void __launch_bounds__(192)
k_gemm_wmma(const float* __restrict__ A, const bf16_t* __restrict__ Wt,
            float* __restrict__ C) {
  __shared__ __align__(32) bf16_t As[M_BLK * K];
  const int tid = threadIdx.x;
  const int mbase = blockIdx.x * M_BLK;

  // cooperative stage: float4 global loads -> packed v4bf LDS stores
  const float4* A4 = (const float4*)(A + (size_t)mbase * K);
  for (int t = tid; t < (M_BLK * K) / 4; t += 192) {
    float4 f = A4[t];
    v4bf o;
    o[0] = (bf16_t)f.x; o[1] = (bf16_t)f.y; o[2] = (bf16_t)f.z; o[3] = (bf16_t)f.w;
    *(v4bf*)(As + t * 4) = o;
  }
  __syncthreads();

  const int wave = tid >> 5;
  const int lane = tid & 31;
  const int lmod = lane & 15;
  const int lhi  = lane >> 4;

  const bf16_t* ap0 = As + lmod * K + lhi * 16;  // rows  0..15
  const bf16_t* ap1 = ap0 + 16 * K;              // rows 16..31
  const bf16_t* wp0 = Wt + (size_t)(wave * 16 + lmod) * K + lhi * 16;
  const bf16_t* wp1 = Wt + (size_t)((wave + 6) * 16 + lmod) * K + lhi * 16;

  v8f acc00 = {}, acc01 = {}, acc10 = {}, acc11 = {};
#pragma unroll 4
  for (int kb = 0; kb < K; kb += 32) {
    v8bf a0lo = *(const v8bf*)(ap0 + kb);
    v8bf a0hi = *(const v8bf*)(ap0 + kb + 8);
    v8bf a1lo = *(const v8bf*)(ap1 + kb);
    v8bf a1hi = *(const v8bf*)(ap1 + kb + 8);
    v16bf af0 = __builtin_shufflevector(a0lo, a0hi,
                  0, 1, 2, 3, 4, 5, 6, 7, 8, 9, 10, 11, 12, 13, 14, 15);
    v16bf af1 = __builtin_shufflevector(a1lo, a1hi,
                  0, 1, 2, 3, 4, 5, 6, 7, 8, 9, 10, 11, 12, 13, 14, 15);
    v16bf bf0 = *(const v16bf*)(wp0 + kb);
    v16bf bf1 = *(const v16bf*)(wp1 + kb);
    acc00 = __builtin_amdgcn_wmma_f32_16x16x32_bf16(false, af0, false, bf0, (short)0, acc00, false, false);
    acc10 = __builtin_amdgcn_wmma_f32_16x16x32_bf16(false, af1, false, bf0, (short)0, acc10, false, false);
    acc01 = __builtin_amdgcn_wmma_f32_16x16x32_bf16(false, af0, false, bf1, (short)0, acc01, false, false);
    acc11 = __builtin_amdgcn_wmma_f32_16x16x32_bf16(false, af1, false, bf1, (short)0, acc11, false, false);
  }

  const int c0 = wave * 16 + lmod;
  const int c1 = (wave + 6) * 16 + lmod;
#pragma unroll
  for (int v = 0; v < 8; ++v) {
    int r0 = mbase + v + 8 * lhi;        // rows  0..15 of the strip
    int r1 = r0 + 16;                    // rows 16..31
    C[(size_t)r0 * NCOLS + c0] = acc00[v];
    C[(size_t)r1 * NCOLS + c0] = acc10[v];
    C[(size_t)r0 * NCOLS + c1] = acc01[v];
    C[(size_t)r1 * NCOLS + c1] = acc11[v];
  }
}

// ---------------------------------------------------------------------------
// Edge scatter for RGCN mean aggregation: 64 threads per edge (one feature
// each). acc[d][r*64+j] += T[s][r*64+j]; cnt[d][r] += 1 (lane j==0).
// ---------------------------------------------------------------------------
__global__ void k_edge_rgcn(const float* __restrict__ T, const int* __restrict__ src,
                            const int* __restrict__ dst, const int* __restrict__ et,
                            float* __restrict__ acc, float* __restrict__ cnt) {
  unsigned gid = blockIdx.x * blockDim.x + threadIdx.x;
  unsigned e = gid >> 6;
  int j = gid & 63;
  if (e >= EE) return;
  int s = src[e], d = dst[e], r = et[e];
  float v = T[(size_t)s * NCOLS + r * 64 + j];
  atomicAdd(acc + (size_t)d * 128 + r * 64 + j, v);
  if (j == 0) atomicAdd(cnt + d * 2 + r, 1.0f);
}

__global__ void k_combine(const float* __restrict__ T, const float* __restrict__ acc,
                          const float* __restrict__ cnt, const float* __restrict__ bias,
                          float* __restrict__ xout, int do_relu) {
  unsigned gid = blockIdx.x * blockDim.x + threadIdx.x;
  unsigned n = gid >> 6;
  int j = gid & 63;
  if (n >= NN) return;
  float c0 = fmaxf(cnt[n * 2 + 0], 1.0f);
  float c1 = fmaxf(cnt[n * 2 + 1], 1.0f);
  float v = T[(size_t)n * NCOLS + 128 + j] + bias[j]
          + acc[(size_t)n * 128 + j] / c0
          + acc[(size_t)n * 128 + 64 + j] / c1;
  if (do_relu) v = fmaxf(v, 0.f);
  xout[(size_t)n * HH + j] = v;
}

// ------------------------------- GAT ---------------------------------------
__global__ void k_gat_node(const float* __restrict__ x2, const float* __restrict__ gw,
                           const float* __restrict__ as_, const float* __restrict__ ad_,
                           float* __restrict__ h, float* __restrict__ emax) {
  int n = blockIdx.x * blockDim.x + threadIdx.x;
  if (n >= NN) return;
  float s = 0.f;
#pragma unroll 8
  for (int j = 0; j < HH; ++j) s += x2[(size_t)n * HH + j] * gw[j];
  h[n] = s;
  emax[n] = leaky02((as_[0] + ad_[0]) * s);   // self-loop score seeds the max
}

__global__ void k_gat_max(const float* __restrict__ h, const int* __restrict__ src,
                          const int* __restrict__ dst, const float* __restrict__ as_,
                          const float* __restrict__ ad_, float* __restrict__ emax) {
  int e = blockIdx.x * blockDim.x + threadIdx.x;
  if (e >= EE) return;
  int s = src[e], d = dst[e];
  float v = leaky02(as_[0] * h[s] + ad_[0] * h[d]);
  atomicMaxFloat(emax + d, v);
}

__global__ void k_gat_init2(const float* __restrict__ h, const float* __restrict__ as_,
                            const float* __restrict__ ad_, const float* __restrict__ emax,
                            float* __restrict__ denom, float* __restrict__ oacc) {
  int n = blockIdx.x * blockDim.x + threadIdx.x;
  if (n >= NN) return;
  float w = __expf(leaky02((as_[0] + ad_[0]) * h[n]) - emax[n]);  // self loop
  denom[n] = w;
  oacc[n] = w * h[n];
}

__global__ void k_gat_edge2(const float* __restrict__ h, const int* __restrict__ src,
                            const int* __restrict__ dst, const float* __restrict__ as_,
                            const float* __restrict__ ad_, const float* __restrict__ emax,
                            float* __restrict__ denom, float* __restrict__ oacc) {
  int e = blockIdx.x * blockDim.x + threadIdx.x;
  if (e >= EE) return;
  int s = src[e], d = dst[e];
  float v = leaky02(as_[0] * h[s] + ad_[0] * h[d]);
  float w = __expf(v - emax[d]);
  atomicAdd(denom + d, w);
  atomicAdd(oacc + d, w * h[s]);
}

__global__ void k_gat_out(const float* __restrict__ oacc, const float* __restrict__ denom,
                          const float* __restrict__ gb, float* __restrict__ out) {
  int n = blockIdx.x * blockDim.x + threadIdx.x;
  if (n >= NN) return;
  out[n] = oacc[n] / denom[n] + gb[0];
}

// ---------------------------------------------------------------------------
extern "C" void kernel_launch(void* const* d_in, const int* in_sizes, int n_in,
                              void* d_out, int out_size, void* d_ws, size_t ws_size,
                              hipStream_t stream) {
  const float* x      = (const float*)d_in[0];
  const int*   ei     = (const int*)d_in[1];
  const int*   et     = (const int*)d_in[2];
  const float* basis1 = (const float*)d_in[3];
  const float* comp1  = (const float*)d_in[4];
  const float* root1  = (const float*)d_in[5];
  const float* bias1  = (const float*)d_in[6];
  const float* basis2 = (const float*)d_in[7];
  const float* comp2  = (const float*)d_in[8];
  const float* root2  = (const float*)d_in[9];
  const float* bias2  = (const float*)d_in[10];
  const float* gat_w  = (const float*)d_in[11];
  const float* att_s  = (const float*)d_in[12];
  const float* att_d  = (const float*)d_in[13];
  const float* gat_b  = (const float*)d_in[14];
  const int* srcp = ei;
  const int* dstp = ei + EE;

  // workspace carving (256B aligned); total ~156 MB
  char* p = (char*)d_ws;
  auto carve = [&](size_t bytes) -> void* {
    void* q = (void*)p;
    p += (bytes + 255) & ~(size_t)255;
    return q;
  };
  bf16_t* Wt1  = (bf16_t*)carve(sizeof(bf16_t) * (size_t)NCOLS * F_IN);
  bf16_t* Wt2  = (bf16_t*)carve(sizeof(bf16_t) * (size_t)NCOLS * HH);
  float* T     = (float*)carve(sizeof(float) * (size_t)NN * NCOLS);
  float* acc   = (float*)carve(sizeof(float) * (size_t)NN * 128);
  float* cnt   = (float*)carve(sizeof(float) * (size_t)NN * 2);
  float* xh    = (float*)carve(sizeof(float) * (size_t)NN * HH);
  float* hvec  = (float*)carve(sizeof(float) * (size_t)NN);
  float* emax  = (float*)carve(sizeof(float) * (size_t)NN);
  float* denom = (float*)carve(sizeof(float) * (size_t)NN);
  float* oacc  = (float*)carve(sizeof(float) * (size_t)NN);
  (void)ws_size; (void)in_sizes; (void)n_in; (void)out_size;

  const int edgeThreads = EE * 64;               // 102.4M, /256 = 400000 blocks

  // ---- weights ----
  k_prep_w<F_IN><<<(NCOLS * F_IN + 255) / 256, 256, 0, stream>>>(basis1, comp1, root1, Wt1);
  k_prep_w<HH>  <<<(NCOLS * HH   + 255) / 256, 256, 0, stream>>>(basis2, comp2, root2, Wt2);

  // ---- RGCN layer 1 ----
  k_gemm_wmma<F_IN><<<NN / M_BLK, 192, 0, stream>>>(x, Wt1, T);
  hipMemsetAsync(acc, 0, sizeof(float) * (size_t)NN * 128, stream);
  hipMemsetAsync(cnt, 0, sizeof(float) * (size_t)NN * 2, stream);
  k_edge_rgcn<<<edgeThreads / 256, 256, 0, stream>>>(T, srcp, dstp, et, acc, cnt);
  k_combine<<<(NN * 64) / 256, 256, 0, stream>>>(T, acc, cnt, bias1, xh, 1);

  // ---- RGCN layer 2 ----
  k_gemm_wmma<HH><<<NN / M_BLK, 192, 0, stream>>>(xh, Wt2, T);
  hipMemsetAsync(acc, 0, sizeof(float) * (size_t)NN * 128, stream);
  hipMemsetAsync(cnt, 0, sizeof(float) * (size_t)NN * 2, stream);
  k_edge_rgcn<<<edgeThreads / 256, 256, 0, stream>>>(T, srcp, dstp, et, acc, cnt);
  k_combine<<<(NN * 64) / 256, 256, 0, stream>>>(T, acc, cnt, bias2, xh, 0);

  // ---- GAT ----
  k_gat_node <<<(NN + 255) / 256, 256, 0, stream>>>(xh, gat_w, att_s, att_d, hvec, emax);
  k_gat_max  <<<(EE + 255) / 256, 256, 0, stream>>>(hvec, srcp, dstp, att_s, att_d, emax);
  k_gat_init2<<<(NN + 255) / 256, 256, 0, stream>>>(hvec, att_s, att_d, emax, denom, oacc);
  k_gat_edge2<<<(EE + 255) / 256, 256, 0, stream>>>(hvec, srcp, dstp, att_s, att_d, emax, denom, oacc);
  k_gat_out  <<<(NN + 255) / 256, 256, 0, stream>>>(oacc, denom, gat_b, (float*)d_out);
}
